// PointTransformerSeg_27436251087208
// MI455X (gfx1250) — matrix operand order repaired and checked by reading, not compiled, not run on hardware
//
#include <hip/hip_runtime.h>
#include <hip/hip_bf16.h>

// Point Transformer segmentation forward for MI455X (gfx1250, wave32).
// Dense linear algebra -> v_wmma_f32_16x16x32_f16 (f16 in, f32 accumulate),
// 128x128 block tiles; A staged row-major, B staged column-major in padded
// LDS so every WMMA fragment is built from contiguous ds_load_b128.
// Irregular ops (KNN/FPS/gather/softmax) are VALU kernels.

#define BATCH 2
#define NPTS  4096
#define NB    4
#define KNN   16
#define DMDL  512

typedef __attribute__((ext_vector_type(16))) _Float16 v16h;
typedef __attribute__((ext_vector_type(8)))  _Float16 v8h;
typedef __attribute__((ext_vector_type(8)))  float    v8f;
typedef __attribute__((ext_vector_type(4)))  float    v4f;

#define CONCAT16(a, b) __builtin_shufflevector(a, b, 0, 1, 2, 3, 4, 5, 6, 7, \
                                               8, 9, 10, 11, 12, 13, 14, 15)

// ---------------------------------------------------------------------------
// WMMA GEMM: C[M,N] = act(A[M,Kd] @ W[Kd,N] + bias + resid)
// 256 threads = 8 waves; block tile 128x128; each wave owns 32x64
// (2x4 WMMA accumulators).
// ---------------------------------------------------------------------------
__global__ __launch_bounds__(256) void gemm_wmma_kernel(
    const float* __restrict__ A, const float* __restrict__ W,
    const float* __restrict__ bias, const float* resid,
    float* C, int M, int Kd, int N, int relu)
{
  __shared__ _Float16 As[128][40];   // [row][k]  (pad 32->40: conflict-free)
  __shared__ _Float16 BsT[128][40];  // [col][k]  column-major B tile
  const int bm = blockIdx.y * 128;
  const int bn = blockIdx.x * 128;
  const int tid  = threadIdx.x;
  const int wave = tid >> 5;
  const int lane = tid & 31;
  const int wm = (wave >> 1) * 32;   // 0,32,64,96
  const int wn = (wave & 1) * 64;    // 0,64
  const int lh = lane >> 4;          // half of wave (0/1)
  const int lr = lane & 15;
  const bool fastA = ((Kd & 31) == 0);

  v8f acc[2][4] = {};

  for (int k0 = 0; k0 < Kd; k0 += 32) {
    // ---- stage A tile (128 x 32): one row-half (16 floats) per thread ----
    {
      int r = tid >> 1, cb = (tid & 1) * 16;
      int gm = bm + r;
      v8h h0 = {}, h1 = {};
      if (gm < M) {
        if (fastA) {                       // aligned full segment (common)
          const v4f* s4 = (const v4f*)(A + (size_t)gm * Kd + k0 + cb);
          v4f f0 = s4[0], f1 = s4[1], f2 = s4[2], f3 = s4[3];
#pragma unroll
          for (int e = 0; e < 4; ++e) {
            h0[e]     = (_Float16)f0[e];
            h0[4 + e] = (_Float16)f1[e];
            h1[e]     = (_Float16)f2[e];
            h1[4 + e] = (_Float16)f3[e];
          }
        } else {       // ragged K: clamped addresses, unconditional loads
          const float* ap = A + (size_t)gm * Kd;
#pragma unroll
          for (int e = 0; e < 8; ++e) {
            int gk = k0 + cb + e;
            float v = ap[(gk < Kd) ? gk : 0];
            h0[e] = (_Float16)((gk < Kd) ? v : 0.0f);
          }
#pragma unroll
          for (int e = 0; e < 8; ++e) {
            int gk = k0 + cb + 8 + e;
            float v = ap[(gk < Kd) ? gk : 0];
            h1[e] = (_Float16)((gk < Kd) ? v : 0.0f);
          }
        }
      }
      *(v8h*)&As[r][cb]     = h0;
      *(v8h*)&As[r][cb + 8] = h1;
    }
    // ---- stage B tile (32 x 128) column-major: one col-half per thread ----
    {
      int col = tid & 127, kb = (tid >> 7) * 16;
      int gn = bn + col;
      v8h h0 = {}, h1 = {};
      if (gn < N) {
        if (k0 + kb + 16 <= Kd) {            // full K segment (common case)
          const float* wp = W + (size_t)(k0 + kb) * N + gn;
#pragma unroll
          for (int e = 0; e < 8; ++e) h0[e] = (_Float16)wp[(size_t)e * N];
#pragma unroll
          for (int e = 0; e < 8; ++e) h1[e] = (_Float16)wp[(size_t)(8 + e) * N];
        } else {       // ragged K: clamped addresses, unconditional loads
#pragma unroll
          for (int e = 0; e < 8; ++e) {
            int gk = k0 + kb + e;
            float v = W[(size_t)((gk < Kd) ? gk : 0) * N + gn];
            h0[e] = (_Float16)((gk < Kd) ? v : 0.0f);
          }
#pragma unroll
          for (int e = 0; e < 8; ++e) {
            int gk = k0 + kb + 8 + e;
            float v = W[(size_t)((gk < Kd) ? gk : 0) * N + gn];
            h1[e] = (_Float16)((gk < Kd) ? v : 0.0f);
          }
        }
      }
      *(v8h*)&BsT[col][kb]     = h0;
      *(v8h*)&BsT[col][kb + 8] = h1;
    }
    // prefetch next K tile into cache while this one is consumed
    if (k0 + 32 < Kd) {
      int ra = bm + (tid >> 1);
      if (ra < M) __builtin_prefetch(A + (size_t)ra * Kd + k0 + 32, 0, 1);
      int rb = k0 + 32 + (tid >> 7) * 16;
      if (rb < Kd) __builtin_prefetch(W + (size_t)rb * N + bn + (tid & 127), 0, 1);
    }
    __syncthreads();

    // fragments: all contiguous 16B LDS loads
    v16h af[2], bf[4];
#pragma unroll
    for (int i = 0; i < 2; ++i) {
      // ISA A 16x32 f16 layout: e0..7 -> k=8*lh.., e8..15 -> k=16+8*lh..
      v8h a0 = *(const v8h*)&As[wm + i * 16 + lr][8 * lh];
      v8h a1 = *(const v8h*)&As[wm + i * 16 + lr][16 + 8 * lh];
      af[i] = CONCAT16(a0, a1);
    }
#pragma unroll
    for (int j = 0; j < 4; ++j) {
      // ISA B 32x16 f16 layout: e0..7 -> k=8*lh.., e8..15 -> k=16+8*lh..
      v8h b0 = *(const v8h*)&BsT[wn + j * 16 + lr][8 * lh];
      v8h b1 = *(const v8h*)&BsT[wn + j * 16 + lr][16 + 8 * lh];
      bf[j] = CONCAT16(b0, b1);
    }
#pragma unroll
    for (int i = 0; i < 2; ++i)
#pragma unroll
      for (int j = 0; j < 4; ++j)
        acc[i][j] = __builtin_amdgcn_wmma_f32_16x16x32_f16(
            false, af[i], false, bf[j], (short)0, acc[i][j], false, false);
    __syncthreads();
  }

#pragma unroll
  for (int i = 0; i < 2; ++i) {
#pragma unroll
    for (int j = 0; j < 4; ++j) {
#pragma unroll
      for (int r = 0; r < 8; ++r) {
        int gm = bm + wm + i * 16 + lh * 8 + r;  // C layout: m = 8*half + vgpr
        int gn = bn + wn + j * 16 + lr;          //            n = lane & 15
        if (gm < M && gn < N) {
          float v = acc[i][j][r];
          if (bias)  v += bias[gn];
          if (resid) v += resid[(size_t)gm * N + gn];
          if (relu)  v = fmaxf(v, 0.0f);
          C[(size_t)gm * N + gn] = v;
        }
      }
    }
  }
}

// ---------------------------------------------------------------------------
// Brute-force KNN (k=16), stable ties like lax.top_k (earliest index first).
// ---------------------------------------------------------------------------
__global__ void knn_kernel(const float* __restrict__ qry,
                           const float* __restrict__ ref,
                           int* __restrict__ out, int nq, int nref)
{
  int t = blockIdx.x * blockDim.x + threadIdx.x;
  if (t >= BATCH * nq) return;
  int b = t / nq;
  const float* qp = qry + (size_t)t * 3;
  float qx = qp[0], qy = qp[1], qz = qp[2];
  const float* rp = ref + (size_t)b * nref * 3;
  float d[KNN]; int id[KNN];
#pragma unroll
  for (int m = 0; m < KNN; ++m) { d[m] = 3.4e38f; id[m] = 0; }
  for (int j = 0; j < nref; ++j) {
    float dx = qx - rp[j * 3 + 0];
    float dy = qy - rp[j * 3 + 1];
    float dz = qz - rp[j * 3 + 2];
    float dd = dx * dx + dy * dy + dz * dz;
    if (dd < d[KNN - 1]) {
      int p = KNN - 1;
      while (p > 0 && dd < d[p - 1]) { d[p] = d[p - 1]; id[p] = id[p - 1]; --p; }
      d[p] = dd; id[p] = j;
    }
  }
  int* op = out + (size_t)t * KNN;
#pragma unroll
  for (int m = 0; m < KNN; ++m) op[m] = id[m];
}

// ---------------------------------------------------------------------------
// Farthest point sampling, one block per batch element, dist array in LDS.
// ---------------------------------------------------------------------------
__global__ __launch_bounds__(256) void fps_kernel(const float* __restrict__ xyz,
                                                  int* __restrict__ out,
                                                  int n, int npoint)
{
  __shared__ float dist[4096];
  __shared__ float rmax[256];
  __shared__ int   ridx[256];
  __shared__ int   far_s;
  int b = blockIdx.x;
  int tid = threadIdx.x;
  const float* p = xyz + (size_t)b * n * 3;
  for (int i = tid; i < n; i += 256) dist[i] = 1e10f;
  if (tid == 0) far_s = 0;
  __syncthreads();
  for (int t = 0; t < npoint; ++t) {
    int far = far_s;
    if (tid == 0) out[b * npoint + t] = far;
    float cx = p[far * 3 + 0], cy = p[far * 3 + 1], cz = p[far * 3 + 2];
    float best = -1.0f; int bi = n;
    for (int i = tid; i < n; i += 256) {
      float dx = p[i * 3 + 0] - cx, dy = p[i * 3 + 1] - cy, dz = p[i * 3 + 2] - cz;
      float dd = dx * dx + dy * dy + dz * dz;
      float nd = fminf(dist[i], dd);
      dist[i] = nd;
      if (nd > best) { best = nd; bi = i; }
    }
    rmax[tid] = best; ridx[tid] = bi;
    __syncthreads();
    for (int s = 128; s > 0; s >>= 1) {
      if (tid < s) {
        if (rmax[tid + s] > rmax[tid] ||
            (rmax[tid + s] == rmax[tid] && ridx[tid + s] < ridx[tid])) {
          rmax[tid] = rmax[tid + s]; ridx[tid] = ridx[tid + s];
        }
      }
      __syncthreads();
    }
    if (tid == 0) far_s = ridx[0];
    __syncthreads();
  }
}

// xyz0[b,i,:] = x[b,i,0:3]   (x has 6 channels)
__global__ void extract_xyz_kernel(const float* __restrict__ x,
                                   float* __restrict__ xyz, int n)
{
  int t = blockIdx.x * blockDim.x + threadIdx.x;
  if (t >= BATCH * n * 3) return;
  int d = t % 3; int row = t / 3;
  xyz[t] = x[(size_t)row * 6 + d];
}

// rel[b,i,k,:] = xyz[b,i,:] - xyz[b,idx[b,i,k],:]
__global__ void rel_kernel(const float* __restrict__ xyz,
                           const int* __restrict__ idx,
                           float* __restrict__ rel, int n)
{
  int t = blockIdx.x * blockDim.x + threadIdx.x;
  if (t >= BATCH * n * KNN) return;
  int pr = t >> 4;          // b*n + i
  int b = pr / n;
  int j = idx[t];
  const float* pi = xyz + (size_t)pr * 3;
  const float* pj = xyz + ((size_t)b * n + j) * 3;
  rel[(size_t)t * 3 + 0] = pi[0] - pj[0];
  rel[(size_t)t * 3 + 1] = pi[1] - pj[1];
  rel[(size_t)t * 3 + 2] = pi[2] - pj[2];
}

// attn_in[b,i,k,c] = q[b,i,c] - xk[b,idx,c] + pos[b,i,k,c]
__global__ void attn_in_kernel(const float* __restrict__ q,
                               const float* __restrict__ xk,
                               const float* __restrict__ pos,
                               const int* __restrict__ idx,
                               float* __restrict__ out, int n)
{
  size_t t = (size_t)blockIdx.x * blockDim.x + threadIdx.x;
  size_t total = (size_t)BATCH * n * KNN * DMDL;
  if (t >= total) return;
  int c = (int)(t & (DMDL - 1));
  size_t row = t >> 9;            // b*n*16 + i*16 + k
  size_t pr  = row >> 4;          // b*n + i
  int b = (int)(pr / n);
  int j = idx[row];
  out[t] = q[pr * DMDL + c] - xk[((size_t)b * n + j) * DMDL + c] + pos[t];
}

// res[b,i,c] = sum_k softmax_k(attn/sqrt(512)) * (v[b,idx,c] + pos[b,i,k,c])
__global__ void attn_agg_kernel(const float* __restrict__ attn,
                                const float* __restrict__ pos,
                                const float* __restrict__ vfeat,
                                const int* __restrict__ idx,
                                float* __restrict__ out, int n)
{
  size_t t = (size_t)blockIdx.x * blockDim.x + threadIdx.x;
  size_t total = (size_t)BATCH * n * DMDL;
  if (t >= total) return;
  int c = (int)(t & (DMDL - 1));
  size_t row = t >> 9;            // b*n + i
  int b = (int)(row / n);
  size_t base = row * KNN;
  const float scale = 0.04419417382415922f;  // 1/sqrt(512)
  float a[KNN]; float mx = -3.4e38f;
#pragma unroll
  for (int k = 0; k < KNN; ++k) {
    a[k] = attn[(base + k) * DMDL + c] * scale;
    mx = fmaxf(mx, a[k]);
  }
  float s = 0.0f;
#pragma unroll
  for (int k = 0; k < KNN; ++k) { a[k] = __expf(a[k] - mx); s += a[k]; }
  float inv = 1.0f / s;
  float r = 0.0f;
#pragma unroll
  for (int k = 0; k < KNN; ++k) {
    int j = idx[base + k];
    r += a[k] * inv * (vfeat[((size_t)b * n + j) * DMDL + c] + pos[(base + k) * DMDL + c]);
  }
  out[row * DMDL + c] = r;
}

// transition-down grouping: g = concat(gather(xyz)-new_xyz, gather(feats))
__global__ void td_concat_kernel(const float* __restrict__ xyz,
                                 const float* __restrict__ nxyz,
                                 const float* __restrict__ feats,
                                 const int* __restrict__ idx,
                                 float* __restrict__ g,
                                 int n, int np, int chin)
{
  int cols = 3 + chin;
  size_t t = (size_t)blockIdx.x * blockDim.x + threadIdx.x;
  size_t total = (size_t)BATCH * np * KNN * cols;
  if (t >= total) return;
  int c = (int)(t % cols);
  size_t row = t / cols;          // b*np*16 + p*16 + k
  size_t pr = row >> 4;           // b*np + p
  int b = (int)(pr / np);
  int j = idx[row];
  float v;
  if (c < 3) v = xyz[((size_t)b * n + j) * 3 + c] - nxyz[pr * 3 + c];
  else       v = feats[((size_t)b * n + j) * chin + (c - 3)];
  g[t] = v;
}

__global__ void maxpool_k_kernel(const float* __restrict__ g,
                                 float* __restrict__ out, int np, int ch)
{
  size_t t = (size_t)blockIdx.x * blockDim.x + threadIdx.x;
  size_t total = (size_t)BATCH * np * ch;
  if (t >= total) return;
  int c = (int)(t % ch);
  size_t row = t / ch;
  float m = -3.4e38f;
#pragma unroll
  for (int k = 0; k < KNN; ++k)
    m = fmaxf(m, g[(row * KNN + k) * (size_t)ch + c]);
  out[t] = m;
}

__global__ void gather3_kernel(const float* __restrict__ xyz,
                               const int* __restrict__ fidx,
                               float* __restrict__ nxyz, int n, int np)
{
  int t = blockIdx.x * blockDim.x + threadIdx.x;
  if (t >= BATCH * np) return;
  int b = t / np;
  int j = fidx[t];
  const float* p = xyz + ((size_t)b * n + j) * 3;
  nxyz[(size_t)t * 3 + 0] = p[0];
  nxyz[(size_t)t * 3 + 1] = p[1];
  nxyz[(size_t)t * 3 + 2] = p[2];
}

// 3-NN of each fine point among coarse points + normalized inverse-dist weights
__global__ void knn3_kernel(const float* __restrict__ fine,
                            const float* __restrict__ coarse,
                            int* __restrict__ id3, float* __restrict__ w3,
                            int n2, int n1)
{
  int t = blockIdx.x * blockDim.x + threadIdx.x;
  if (t >= BATCH * n2) return;
  int b = t / n2;
  const float* fp = fine + (size_t)t * 3;
  float fx = fp[0], fy = fp[1], fz = fp[2];
  const float* cp = coarse + (size_t)b * n1 * 3;
  float d0 = 3.4e38f, d1 = 3.4e38f, d2 = 3.4e38f;
  int i0 = 0, i1 = 0, i2 = 0;
  for (int j = 0; j < n1; ++j) {
    float dx = fx - cp[j * 3 + 0], dy = fy - cp[j * 3 + 1], dz = fz - cp[j * 3 + 2];
    float dd = dx * dx + dy * dy + dz * dz;
    if (dd < d0)      { d2 = d1; i2 = i1; d1 = d0; i1 = i0; d0 = dd; i0 = j; }
    else if (dd < d1) { d2 = d1; i2 = i1; d1 = dd; i1 = j; }
    else if (dd < d2) { d2 = dd; i2 = j; }
  }
  float w0 = 1.0f / (fmaxf(d0, 0.0f) + 1e-8f);
  float w1 = 1.0f / (fmaxf(d1, 0.0f) + 1e-8f);
  float w2 = 1.0f / (fmaxf(d2, 0.0f) + 1e-8f);
  float s = w0 + w1 + w2;
  id3[t * 3 + 0] = i0; id3[t * 3 + 1] = i1; id3[t * 3 + 2] = i2;
  w3[t * 3 + 0] = w0 / s; w3[t * 3 + 1] = w1 / s; w3[t * 3 + 2] = w2 / s;
}

__global__ void interp_add_kernel(const float* __restrict__ f1,
                                  const float* __restrict__ f2,
                                  const int* __restrict__ id3,
                                  const float* __restrict__ w3,
                                  float* __restrict__ out,
                                  int n1, int n2, int ch)
{
  size_t t = (size_t)blockIdx.x * blockDim.x + threadIdx.x;
  size_t total = (size_t)BATCH * n2 * ch;
  if (t >= total) return;
  int c = (int)(t % ch);
  size_t row = t / ch;          // b*n2 + j
  int b = (int)(row / n2);
  float acc = f2[t];
#pragma unroll
  for (int m = 0; m < 3; ++m) {
    int j = id3[row * 3 + m];
    acc += w3[row * 3 + m] * f1[((size_t)b * n1 + j) * ch + c];
  }
  out[t] = acc;
}

// ---------------------------------------------------------------------------
// Host orchestration
// ---------------------------------------------------------------------------
struct Lin { const float* w; const float* b; };
struct TBlock {
  Lin fc1; const float* wq; const float* wk; const float* wv;
  Lin d0, d1, g0, g1, fc2;
};

static inline unsigned gsz(size_t n) { return (unsigned)((n + 255) / 256); }

static void gemm(hipStream_t s, const float* A, const float* W, const float* bias,
                 const float* resid, float* C, int M, int Kd, int N, bool relu)
{
  dim3 grid((unsigned)((N + 127) / 128), (unsigned)((M + 127) / 128));
  gemm_wmma_kernel<<<grid, 256, 0, s>>>(A, W, bias, resid, C, M, Kd, N, relu ? 1 : 0);
}

static Lin rdlin(void* const* d_in, int& ii) {
  Lin l; l.w = (const float*)d_in[ii++]; l.b = (const float*)d_in[ii++]; return l;
}
static TBlock rdblk(void* const* d_in, int& ii) {
  TBlock t;
  t.fc1 = rdlin(d_in, ii);
  t.wq = (const float*)d_in[ii++];
  t.wk = (const float*)d_in[ii++];
  t.wv = (const float*)d_in[ii++];
  t.d0 = rdlin(d_in, ii); t.d1 = rdlin(d_in, ii);
  t.g0 = rdlin(d_in, ii); t.g1 = rdlin(d_in, ii);
  t.fc2 = rdlin(d_in, ii);
  return t;
}

struct Arena {
  char* base; size_t cap; size_t off;
  void* take(size_t bytes) {
    size_t a = (off + 255) & ~(size_t)255;
    off = a + bytes;
    return (void*)(base + a);
  }
  float* f(size_t n) { return (float*)take(n * 4); }
  int*   i(size_t n) { return (int*)take(n * 4); }
};

static void run_transformer(hipStream_t s, const TBlock& p, int dp,
                            const float* xyz, float* feats, int n,
                            int* idx, float* X, float* Q, float* Kb, float* Vb,
                            float* R, float* B1, float* B2, float* B3)
{
  int rows = BATCH * n;
  size_t krows = (size_t)rows * KNN;
  knn_kernel<<<gsz(rows), 256, 0, s>>>(xyz, xyz, idx, n, n);
  gemm(s, feats, p.fc1.w, p.fc1.b, nullptr, X, rows, dp, DMDL, false);
  gemm(s, X, p.wq, nullptr, nullptr, Q,  rows, DMDL, DMDL, false);
  gemm(s, X, p.wk, nullptr, nullptr, Kb, rows, DMDL, DMDL, false);
  gemm(s, X, p.wv, nullptr, nullptr, Vb, rows, DMDL, DMDL, false);
  rel_kernel<<<gsz(krows), 256, 0, s>>>(xyz, idx, R, n);
  gemm(s, R,  p.d0.w, p.d0.b, nullptr, B1, (int)krows, 3,    DMDL, true);
  gemm(s, B1, p.d1.w, p.d1.b, nullptr, B2, (int)krows, DMDL, DMDL, false);   // pos
  attn_in_kernel<<<gsz(krows * DMDL), 256, 0, s>>>(Q, Kb, B2, idx, B1, n);
  gemm(s, B1, p.g0.w, p.g0.b, nullptr, B3, (int)krows, DMDL, DMDL, true);
  gemm(s, B3, p.g1.w, p.g1.b, nullptr, B1, (int)krows, DMDL, DMDL, false);   // attn
  attn_agg_kernel<<<gsz((size_t)rows * DMDL), 256, 0, s>>>(B1, B2, Vb, idx, X, n);
  gemm(s, X, p.fc2.w, p.fc2.b, feats, feats, rows, DMDL, dp, false);         // +resid
}

extern "C" void kernel_launch(void* const* d_in, const int* in_sizes, int n_in,
                              void* d_out, int out_size, void* d_ws, size_t ws_size,
                              hipStream_t stream)
{
  (void)in_sizes; (void)n_in; (void)out_size;
  int ii = 0;
  const float* x = (const float*)d_in[ii++];
  Lin fc1a = rdlin(d_in, ii), fc1b = rdlin(d_in, ii);
  TBlock tr1 = rdblk(d_in, ii);
  Lin tdp[NB][2];
  for (int i = 0; i < NB; ++i) { tdp[i][0] = rdlin(d_in, ii); tdp[i][1] = rdlin(d_in, ii); }
  TBlock tre[NB];
  for (int i = 0; i < NB; ++i) tre[i] = rdblk(d_in, ii);
  Lin fc2a = rdlin(d_in, ii), fc2b = rdlin(d_in, ii), fc2c = rdlin(d_in, ii);
  TBlock tr2 = rdblk(d_in, ii);
  Lin tup[NB][2];
  for (int i = 0; i < NB; ++i) { tup[i][0] = rdlin(d_in, ii); tup[i][1] = rdlin(d_in, ii); }
  TBlock trd[NB];
  for (int i = 0; i < NB; ++i) trd[i] = rdblk(d_in, ii);
  Lin fc3l[4];
  for (int i = 0; i < 4; ++i) fc3l[i] = rdlin(d_in, ii);

  static const int n_l[5]  = {4096, 1024, 256, 64, 16};
  static const int ch_l[5] = {32, 64, 128, 256, 512};

  Arena ar{(char*)d_ws, ws_size, 0};
  float* xyzL[5]; float* fe[5]; float* fd[5];
  for (int l = 0; l < 5; ++l) xyzL[l] = ar.f((size_t)BATCH * n_l[l] * 3);
  for (int l = 0; l < 5; ++l) fe[l]   = ar.f((size_t)BATCH * n_l[l] * ch_l[l]);
  for (int l = 0; l < 5; ++l) fd[l]   = ar.f((size_t)BATCH * n_l[l] * ch_l[l]);
  int*   fidx = ar.i((size_t)BATCH * 1024);
  int*   idx  = ar.i((size_t)BATCH * NPTS * KNN);
  float* X  = ar.f((size_t)BATCH * NPTS * DMDL);
  float* Q  = ar.f((size_t)BATCH * NPTS * DMDL);
  float* Kb = ar.f((size_t)BATCH * NPTS * DMDL);
  float* Vb = ar.f((size_t)BATCH * NPTS * DMDL);
  float* R  = ar.f((size_t)BATCH * NPTS * KNN * 3);
  float* B1 = ar.f((size_t)BATCH * NPTS * KNN * DMDL);
  float* B2 = ar.f((size_t)BATCH * NPTS * KNN * DMDL);
  float* B3 = ar.f((size_t)BATCH * NPTS * KNN * DMDL);
  float* T32 = ar.f((size_t)BATCH * NPTS * 32);
  float* Ta  = ar.f((size_t)BATCH * 16 * DMDL);
  float* Tb  = ar.f((size_t)BATCH * 16 * DMDL);
  float* H1 = ar.f((size_t)BATCH * NPTS * 64);
  float* H2 = ar.f((size_t)BATCH * NPTS * 128);
  float* H3 = ar.f((size_t)BATCH * NPTS * 256);
  if (ar.off > ws_size) return;   // scratch insufficient — bail deterministically

  hipStream_t s = stream;
  float* out = (float*)d_out;

  // xyz level 0 + input MLP
  extract_xyz_kernel<<<gsz((size_t)BATCH * NPTS * 3), 256, 0, s>>>(x, xyzL[0], NPTS);
  gemm(s, x, fc1a.w, fc1a.b, nullptr, T32, BATCH * NPTS, 6, 32, true);
  gemm(s, T32, fc1b.w, fc1b.b, nullptr, fe[0], BATCH * NPTS, 32, 32, false);
  run_transformer(s, tr1, 32, xyzL[0], fe[0], NPTS, idx, X, Q, Kb, Vb, R, B1, B2, B3);

  // encoder
  for (int i = 0; i < NB; ++i) {
    int n = n_l[i], np = n_l[i + 1], chin = ch_l[i], ch = ch_l[i + 1];
    fps_kernel<<<BATCH, 256, 0, s>>>(xyzL[i], fidx, n, np);
    gather3_kernel<<<gsz((size_t)BATCH * np), 256, 0, s>>>(xyzL[i], fidx, xyzL[i + 1], n, np);
    knn_kernel<<<gsz((size_t)BATCH * np), 256, 0, s>>>(xyzL[i + 1], xyzL[i], idx, np, n);
    size_t grows = (size_t)BATCH * np * KNN;
    td_concat_kernel<<<gsz(grows * (3 + chin)), 256, 0, s>>>(
        xyzL[i], xyzL[i + 1], fe[i], idx, B1, n, np, chin);
    gemm(s, B1, tdp[i][0].w, tdp[i][0].b, nullptr, B2, (int)grows, 3 + chin, ch, true);
    gemm(s, B2, tdp[i][1].w, tdp[i][1].b, nullptr, B1, (int)grows, ch, ch, true);
    maxpool_k_kernel<<<gsz((size_t)BATCH * np * ch), 256, 0, s>>>(B1, fe[i + 1], np, ch);
    run_transformer(s, tre[i], ch, xyzL[i + 1], fe[i + 1], np, idx, X, Q, Kb, Vb, R, B1, B2, B3);
  }

  // bottleneck MLP + transformer
  gemm(s, fe[4], fc2a.w, fc2a.b, nullptr, Ta, BATCH * 16, 512, 512, true);
  gemm(s, Ta, fc2b.w, fc2b.b, nullptr, Tb, BATCH * 16, 512, 512, true);
  gemm(s, Tb, fc2c.w, fc2c.b, nullptr, fd[4], BATCH * 16, 512, 512, false);
  run_transformer(s, tr2, 512, xyzL[4], fd[4], 16, idx, X, Q, Kb, Vb, R, B1, B2, B3);

  // decoder
  for (int i = 0; i < NB; ++i) {
    int lc = 4 - i, lf = 3 - i;
    int n1 = n_l[lc], n2 = n_l[lf], ch = ch_l[lf];
    gemm(s, fd[lc], tup[i][0].w, tup[i][0].b, nullptr, X, BATCH * n1, 2 * ch, ch, true);
    gemm(s, fe[lf], tup[i][1].w, tup[i][1].b, nullptr, Q, BATCH * n2, ch, ch, true);
    knn3_kernel<<<gsz((size_t)BATCH * n2), 256, 0, s>>>(xyzL[lf], xyzL[lc], idx, R, n2, n1);
    interp_add_kernel<<<gsz((size_t)BATCH * n2 * ch), 256, 0, s>>>(
        X, Q, idx, R, fd[lf], n1, n2, ch);
    run_transformer(s, trd[i], ch, xyzL[lf], fd[lf], n2, idx, X, Q, Kb, Vb, R, B1, B2, B3);
  }

  // head
  gemm(s, fd[0], fc3l[0].w, fc3l[0].b, nullptr, H1, BATCH * NPTS, 32, 64, true);
  gemm(s, H1, fc3l[1].w, fc3l[1].b, nullptr, H2, BATCH * NPTS, 64, 128, true);
  gemm(s, H2, fc3l[2].w, fc3l[2].b, nullptr, H3, BATCH * NPTS, 128, 256, true);
  gemm(s, H3, fc3l[3].w, fc3l[3].b, nullptr, out, BATCH * NPTS, 256, 512, false);
}
